// ConfigurableGNN_37984690766193
// MI455X (gfx1250) — matrix operand order repaired and checked by reading, not compiled, not run on hardware
//
#include <hip/hip_runtime.h>
#include <hip/hip_bf16.h>

typedef __attribute__((ext_vector_type(16))) __bf16 v16bf;
typedef __attribute__((ext_vector_type(8)))  float  v8f;

#define HDIM 256

// ---------------------------------------------------------------------------
// Pack W[256,256] (f32, row-major) into WMMA B-fragment order, bf16 hi/lo.
// Fragment index: frag = (colTile*8 + kChunk)*32 + lane, 16 bf16 per frag.
// B layout (16-bit, 32x16): lane -> n = colTile*16 + (lane&15);
//   K = kChunk*32 + 16*(lane>>4) + 2r + {0,1}  -> packed positions 2r, 2r+1.
// 4096 threads total (16 colTiles * 8 kChunks * 32 lanes).
// ---------------------------------------------------------------------------
__global__ __launch_bounds__(256) void k_packW(const float* __restrict__ W,
                                               __bf16* __restrict__ ph,
                                               __bf16* __restrict__ pl) {
  const int t = blockIdx.x * blockDim.x + threadIdx.x;   // == frag index
  const int lane = t & 31;
  const int c    = (t >> 5) & 7;
  const int ct   = t >> 8;
  const int n    = (ct << 4) + (lane & 15);
  const int kb   = (c << 5) + ((lane >> 4) << 4);
  __bf16* dh = ph + (size_t)t * 16;
  __bf16* dl = pl + (size_t)t * 16;
#pragma unroll
  for (int r = 0; r < 8; ++r) {
    const int k = kb + (r << 1);
    const float y0 = W[(size_t)k * HDIM + n];
    const float y1 = W[(size_t)(k + 1) * HDIM + n];
    const __bf16 h0 = (__bf16)y0, h1 = (__bf16)y1;
    dh[2 * r]     = h0;
    dh[2 * r + 1] = h1;
    dl[2 * r]     = (__bf16)(y0 - (float)h0);
    dl[2 * r + 1] = (__bf16)(y1 - (float)h1);
  }
}

// ---------------------------------------------------------------------------
// GEMM: C[N,256] = A[N,256] @ W[256,256] with pre-packed bf16 hi/lo W.
// Block = 256 threads = 8 waves, owns a 16-row strip. Staging converts the
// f32 A tile into packed bf16 hi/lo fragments in LDS (one cvt per element);
// compute loop is pure b128 loads + 3 bf16-split WMMAs per 16x16 tile chunk.
// A layout (16-bit, 16x32): lane -> m = lane&15;
//   K = kChunk*32 + 16*(lane>>4) + 2r + {0,1}  -> packed positions 2r, 2r+1.
// ---------------------------------------------------------------------------
__global__ __launch_bounds__(256) void k_gemm_bf16x3(
    const float* __restrict__ A,
    const __bf16* __restrict__ Wph, const __bf16* __restrict__ Wpl,
    float* __restrict__ C, int N)
{
  __shared__ __align__(32) __bf16 sAhi[8 * 32 * 16];   // 8 KB
  __shared__ __align__(32) __bf16 sAlo[8 * 32 * 16];   // 8 KB
  const int row0 = blockIdx.x << 4;
  const int t = threadIdx.x;

  { // stage + convert: thread t owns fragment (chunk c, lane l)
    const int c = t >> 5;
    const int l = t & 31;
    const int m = l & 15;
    const int half = l >> 4;
    const int row = row0 + m;
    __bf16* dh = &sAhi[(size_t)t * 16];
    __bf16* dl = &sAlo[(size_t)t * 16];
    if (row < N) {
      const float* sp = A + (size_t)row * HDIM;
#pragma unroll
      for (int r = 0; r < 8; ++r) {
        const int kb = (c << 5) + (half << 4) + (r << 1);
        const float x0 = sp[kb];
        const float x1 = sp[kb + 1];
        const __bf16 h0 = (__bf16)x0, h1 = (__bf16)x1;
        dh[2 * r]     = h0;
        dh[2 * r + 1] = h1;
        dl[2 * r]     = (__bf16)(x0 - (float)h0);
        dl[2 * r + 1] = (__bf16)(x1 - (float)h1);
      }
    } else {
#pragma unroll
      for (int r = 0; r < 16; ++r) { dh[r] = (__bf16)0.f; dl[r] = (__bf16)0.f; }
    }
  }
  __syncthreads();

  const int wave = t >> 5;
  const int lane = t & 31;
  const int half = lane >> 4;
  const int m    = lane & 15;

  v8f acc0 = {};
  v8f acc1 = {};

  for (int c = 0; c < 8; ++c) {
    const v16bf ahi = *(const v16bf*)&sAhi[(size_t)((c << 5) + lane) * 16];
    const v16bf alo = *(const v16bf*)&sAlo[(size_t)((c << 5) + lane) * 16];
#pragma unroll
    for (int tt = 0; tt < 2; ++tt) {
      const int ct = (wave << 1) + tt;
      const size_t fi = (size_t)(((ct << 3) + c) << 5 | lane) * 16;
      const v16bf bhi = *(const v16bf*)(Wph + fi);
      const v16bf blo = *(const v16bf*)(Wpl + fi);
      v8f acc = tt ? acc1 : acc0;
      acc = __builtin_amdgcn_wmma_f32_16x16x32_bf16(false, ahi, false, bhi, (short)0, acc, false, false);
      acc = __builtin_amdgcn_wmma_f32_16x16x32_bf16(false, alo, false, bhi, (short)0, acc, false, false);
      acc = __builtin_amdgcn_wmma_f32_16x16x32_bf16(false, ahi, false, blo, (short)0, acc, false, false);
      if (tt) acc1 = acc; else acc0 = acc;
    }
  }

  // C/D layout: VGPR r -> M = r + 8*half, N = colTile*16 + (lane&15)
#pragma unroll
  for (int tt = 0; tt < 2; ++tt) {
    const int n0 = ((wave << 1) + tt) << 4;
    const v8f acc = tt ? acc1 : acc0;
#pragma unroll
    for (int r = 0; r < 8; ++r) {
      const int row = row0 + r + (half << 3);
      if (row < N) C[(size_t)row * HDIM + n0 + m] = acc[r];
    }
  }
}

// ---------------------------------------------------------------------------
// Small utility kernels
// ---------------------------------------------------------------------------
__global__ void k_fill(float* __restrict__ p, float v, int n) {
  int i = blockIdx.x * blockDim.x + threadIdx.x;
  if (i < n) p[i] = v;
}

__global__ void k_deg(const int* __restrict__ dst, float* __restrict__ deg, int E) {
  int e = blockIdx.x * blockDim.x + threadIdx.x;
  if (e < E) atomicAdd(&deg[dst[e]], 1.0f);
}

__global__ void k_rsqrt_inplace(float* __restrict__ p, int n) {
  int i = blockIdx.x * blockDim.x + threadIdx.x;
  if (i < n) p[i] = rsqrtf(p[i]);   // deg >= 1 always (self-loop)
}

__global__ void k_wnorm(const int* __restrict__ src, const int* __restrict__ dst,
                        const float* __restrict__ dinv, float* __restrict__ wnorm, int E) {
  int e = blockIdx.x * blockDim.x + threadIdx.x;
  if (e < E) wnorm[e] = dinv[src[e]] * dinv[dst[e]];
}

// edge scatter: agg[dst] += hw[src] * wnorm[e]; 64 threads per edge, float4 gather
__global__ void k_edge_agg(const int* __restrict__ src, const int* __restrict__ dst,
                           const float* __restrict__ hw, const float* __restrict__ wnorm,
                           float* __restrict__ agg, int total) {
  int idx = blockIdx.x * blockDim.x + threadIdx.x;
  if (idx >= total) return;
  const int e  = idx >> 6;
  const int fo = (idx & 63) << 2;
  const int s = src[e];
  const int d = dst[e];
  const float w = wnorm[e];
  const float4 v = *(const float4*)(hw + (size_t)s * HDIM + fo);
  float* p = agg + (size_t)d * HDIM + fo;
  atomicAdd(p + 0, v.x * w);
  atomicAdd(p + 1, v.y * w);
  atomicAdd(p + 2, v.z * w);
  atomicAdd(p + 3, v.w * w);
}

// hout = relu(agg + hw*dinv^2 + b)   (self-loop + bias + activation)
__global__ void k_self_relu(const float* __restrict__ agg, const float* __restrict__ hw,
                            const float* __restrict__ dinv, const float* __restrict__ b,
                            float* __restrict__ hout, int total) {
  int idx = blockIdx.x * blockDim.x + threadIdx.x;
  if (idx >= total) return;
  const int i = idx >> 8;
  const int f = idx & 255;
  const float di = dinv[i];
  const float v = agg[idx] + hw[idx] * di * di + b[f];
  hout[idx] = fmaxf(v, 0.f);
}

// sorted-batch mean-pool partials: run-length accumulate, flush on graph change
__global__ __launch_bounds__(256) void k_pool(const float* __restrict__ h,
                                              const int* __restrict__ batch,
                                              float* __restrict__ pooled,
                                              float* __restrict__ cnt, int N) {
  const int f = threadIdx.x;
  const int start = blockIdx.x * 256;
  if (start >= N) return;
  const int end = min(start + 256, N);
  int curg = batch[start];
  float accum = 0.f, c = 0.f;
  for (int i = start; i < end; ++i) {
    const int g = batch[i];
    if (g != curg) {
      atomicAdd(&pooled[(size_t)curg * HDIM + f], accum);
      if (f == 0) atomicAdd(&cnt[curg], c);
      accum = 0.f; c = 0.f; curg = g;
    }
    accum += h[(size_t)i * HDIM + f];
    c += 1.f;
  }
  atomicAdd(&pooled[(size_t)curg * HDIM + f], accum);
  if (f == 0) atomicAdd(&cnt[curg], c);
}

// classifier head: out[g] = relu(mean_g @ cW1 + cb1) @ cW2 + cb2
__global__ __launch_bounds__(128) void k_classifier(
    const float* __restrict__ pooled, const float* __restrict__ cnt,
    const float* __restrict__ cW1, const float* __restrict__ cb1,
    const float* __restrict__ cW2, const float* __restrict__ cb2,
    float* __restrict__ out) {
  __shared__ float z[128];
  const int g = blockIdx.x;
  const int j = threadIdx.x;
  const float inv = 1.0f / fmaxf(cnt[g], 1.0f);
  float acc = cb1[j];
  for (int k = 0; k < HDIM; ++k)
    acc += (pooled[(size_t)g * HDIM + k] * inv) * cW1[k * 128 + j];
  z[j] = fmaxf(acc, 0.f);
  __syncthreads();
  if (j < 16) {
    float a2 = cb2[j];
    for (int k = 0; k < 128; ++k) a2 += z[k] * cW2[k * 16 + j];
    out[g * 16 + j] = a2;
  }
}

// ---------------------------------------------------------------------------
extern "C" void kernel_launch(void* const* d_in, const int* in_sizes, int n_in,
                              void* d_out, int out_size, void* d_ws, size_t ws_size,
                              hipStream_t stream) {
  const float* x     = (const float*)d_in[0];
  const int*   ei    = (const int*)  d_in[1];
  const int*   batch = (const int*)  d_in[2];
  const float* W0 = (const float*)d_in[3];  const float* b0 = (const float*)d_in[4];
  const float* W1 = (const float*)d_in[5];  const float* b1 = (const float*)d_in[6];
  const float* W2 = (const float*)d_in[7];  const float* b2 = (const float*)d_in[8];
  const float* cW1 = (const float*)d_in[9];  const float* cb1 = (const float*)d_in[10];
  const float* cW2 = (const float*)d_in[11]; const float* cb2 = (const float*)d_in[12];
  float* out = (float*)d_out;

  const int E = in_sizes[1] / 2;
  const int N = in_sizes[2];
  const int G = out_size / 16;
  const int* src = ei;
  const int* dst = ei + E;

  // workspace carve-up (all 64-float = 256B aligned)
  float* ws = (float*)d_ws;
  size_t off = 0;
  auto al = [](size_t v) { return (v + 63) & ~(size_t)63; };
  float* deg    = ws + off; off += al((size_t)N);          // becomes dinv in-place
  float* wnorm  = ws + off; off += al((size_t)E);
  float* hw     = ws + off; off += (size_t)N * HDIM;
  float* agg    = ws + off; off += (size_t)N * HDIM;
  float* hbuf   = ws + off; off += (size_t)N * HDIM;
  float* pooled = ws + off; off += (size_t)G * HDIM;
  float* cnt    = ws + off; off += al((size_t)G);
  __bf16* wph   = (__bf16*)(ws + off); off += (size_t)HDIM * HDIM / 2;  // 65536 bf16
  __bf16* wpl   = (__bf16*)(ws + off); off += (size_t)HDIM * HDIM / 2;
  (void)ws_size; (void)n_in;

  const int TPB = 256;
  const int NH  = N * HDIM;

  // degrees (self-loop counts as 1) and normalization
  k_fill<<<(N + TPB - 1) / TPB, TPB, 0, stream>>>(deg, 1.0f, N);
  k_deg<<<(E + TPB - 1) / TPB, TPB, 0, stream>>>(dst, deg, E);
  k_rsqrt_inplace<<<(N + TPB - 1) / TPB, TPB, 0, stream>>>(deg, N);
  k_wnorm<<<(E + TPB - 1) / TPB, TPB, 0, stream>>>(src, dst, deg, wnorm, E);

  const float* Ws[3] = {W0, W1, W2};
  const float* bs[3] = {b0, b1, b2};
  const float* hin = x;
  for (int l = 0; l < 3; ++l) {
    k_packW<<<16, 256, 0, stream>>>(Ws[l], wph, wpl);
    k_gemm_bf16x3<<<(N + 15) / 16, 256, 0, stream>>>(hin, wph, wpl, hw, N);
    k_fill<<<(NH + TPB - 1) / TPB, TPB, 0, stream>>>(agg, 0.0f, NH);
    const int etotal = E * 64;
    k_edge_agg<<<(etotal + TPB - 1) / TPB, TPB, 0, stream>>>(src, dst, hw, wnorm, agg, etotal);
    k_self_relu<<<(NH + TPB - 1) / TPB, TPB, 0, stream>>>(agg, hw, deg, bs[l], hbuf, NH);
    hin = hbuf;
  }

  // mean pool + head (pooled and cnt are contiguous -> one fill)
  k_fill<<<(G * HDIM + G + TPB - 1) / TPB, TPB, 0, stream>>>(pooled, 0.0f, G * HDIM + G);
  k_pool<<<(N + 255) / 256, 256, 0, stream>>>(hbuf, batch, pooled, cnt, N);
  k_classifier<<<G, 128, 0, stream>>>(pooled, cnt, cW1, cb1, cW2, cb2, out);
}
